// CMINVideoEncoder_68247030334355
// MI455X (gfx1250) — compile-verified, matrix operand order checked by reading, not run on hardware
//
#include <hip/hip_runtime.h>
#include <cstddef>

// ---------------------------------------------------------------------------
// CDNA5 (gfx1250) implementation: bf16 WMMA GEMMs + banded attention + BiGRU.
// wave32 throughout. v_wmma_f32_16x16x32_bf16 is the workhorse.
// ---------------------------------------------------------------------------

#define BB   64
#define TT   256
#define DD   1024
#define HH   8
#define DK   128
#define HID  512
#define GHID 256
#define MTOT (BB * TT)          // 16384 rows in the residual stream

typedef __attribute__((ext_vector_type(16))) __bf16 v16bf;
typedef __attribute__((ext_vector_type(8)))  float  v8f;
typedef __attribute__((ext_vector_type(4)))  unsigned int u32x4;

union FragU { v16bf v; u32x4 u[2]; };

__device__ __forceinline__ unsigned short f2bf(float f) {
    unsigned u = __float_as_uint(f);
    unsigned r = u + 0x7FFFu + ((u >> 16) & 1u);   // round-to-nearest-even
    return (unsigned short)(r >> 16);
}

// Load a 16x32 bf16 WMMA fragment (A or B) from row-major [rows, K] memory.
// Lane l (l&15 = row/col), hi half holds K {8..15, 24..31}: two 16B loads.
__device__ __forceinline__ v16bf load_frag_g(const unsigned short* __restrict__ p) {
    FragU f;
    f.u[0] = *(const u32x4*)(p);        // K = kc .. kc+7
    f.u[1] = *(const u32x4*)(p + 16);   // K = kc+16 .. kc+23
    return f.v;
}

// ---------------------------------------------------------------------------
// f32 -> bf16 conversion (grid-stride, 4-wide)
// ---------------------------------------------------------------------------
__global__ __launch_bounds__(256)
void cvt_f32_bf16_kernel(const float* __restrict__ in, unsigned short* __restrict__ out, int n4) {
    // n4 = number of float4 groups (element count / 4; all our tensors are /4)
    for (unsigned i = blockIdx.x * 256 + threadIdx.x; i < (unsigned)n4; i += gridDim.x * 256) {
        const float4 v = *(const float4*)(in + (size_t)i * 4);
        uint2 pk;
        pk.x = (unsigned)f2bf(v.x) | ((unsigned)f2bf(v.y) << 16);
        pk.y = (unsigned)f2bf(v.z) | ((unsigned)f2bf(v.w) << 16);
        *(uint2*)(out + (size_t)i * 4) = pk;
    }
}

__global__ __launch_bounds__(256)
void zero_f32_kernel(float* __restrict__ p, int n) {
    for (unsigned i = blockIdx.x * 256 + threadIdx.x; i < (unsigned)n; i += gridDim.x * 256)
        p[i] = 0.0f;
}

// ---------------------------------------------------------------------------
// bf16 GEMM: out[M,N] = A[M,K](bf16) * W[N,K](bf16)^T + bias[N] (+ resid[M,N])
// 256 threads = 8 waves; block tile 128x128; wave tile 32x64 (2x4 WMMA tiles).
// Requires M%128==0, N%128==0, K%32==0. All flat offsets fit in 32 bits.
// ---------------------------------------------------------------------------
template <bool HAS_RESID>
__global__ __launch_bounds__(256)
void gemm_bf16_kernel(const unsigned short* __restrict__ A,
                      const unsigned short* __restrict__ W,
                      const float* __restrict__ bias,
                      const float* __restrict__ resid,
                      float* __restrict__ out,
                      int M, int N, int K)
{
    const unsigned tid  = threadIdx.x;
    const unsigned w    = tid >> 5;
    const unsigned lane = tid & 31;
    const unsigned lr   = lane & 15;
    const unsigned hi   = lane >> 4;
    const unsigned mw   = w >> 1;            // 0..3
    const unsigned nw   = w & 1;             // 0..1
    const unsigned mbase = blockIdx.y * 128 + mw * 32;
    const unsigned nbase = blockIdx.x * 128 + nw * 64;

    v8f acc[2][4];
#pragma unroll
    for (int i = 0; i < 2; ++i)
#pragma unroll
        for (int j = 0; j < 4; ++j)
            acc[i][j] = (v8f){0.f,0.f,0.f,0.f,0.f,0.f,0.f,0.f};

    // hoisted fragment base pointers (advance by 32 elements per K-step)
    const unsigned short* ap0 = A + (mbase + 0 * 16 + lr) * (unsigned)K + hi * 8;
    const unsigned short* ap1 = A + (mbase + 1 * 16 + lr) * (unsigned)K + hi * 8;
    const unsigned short* wp0 = W + (nbase + 0 * 16 + lr) * (unsigned)K + hi * 8;
    const unsigned short* wp1 = W + (nbase + 1 * 16 + lr) * (unsigned)K + hi * 8;
    const unsigned short* wp2 = W + (nbase + 2 * 16 + lr) * (unsigned)K + hi * 8;
    const unsigned short* wp3 = W + (nbase + 3 * 16 + lr) * (unsigned)K + hi * 8;

    for (int k0 = 0; k0 < K; k0 += 32) {
        v16bf a[2], b[4];
        a[0] = load_frag_g(ap0); ap0 += 32;
        a[1] = load_frag_g(ap1); ap1 += 32;
        b[0] = load_frag_g(wp0); wp0 += 32;
        b[1] = load_frag_g(wp1); wp1 += 32;
        b[2] = load_frag_g(wp2); wp2 += 32;
        b[3] = load_frag_g(wp3); wp3 += 32;
#pragma unroll
        for (int i = 0; i < 2; ++i)
#pragma unroll
            for (int j = 0; j < 4; ++j)
                acc[i][j] = __builtin_amdgcn_wmma_f32_16x16x32_bf16(
                    false, a[i], false, b[j], (short)0, acc[i][j], false, false);
    }

    // C layout: VGPR r, lanes 0-15: (M=r, N=lane); lanes 16-31: (M=8+r, N=lane-16)
#pragma unroll
    for (int i = 0; i < 2; ++i) {
#pragma unroll
        for (int j = 0; j < 4; ++j) {
            const unsigned col = nbase + j * 16 + lr;
            const float bv = bias[col];
            const unsigned base = (mbase + i * 16 + hi * 8) * (unsigned)N + col;
            float* op = out + base;
            const float* rp = resid + base;   // unused when !HAS_RESID
#pragma unroll
            for (int r = 0; r < 8; ++r) {
                float v = acc[i][j][r] + bv;
                if (HAS_RESID) v += rp[(unsigned)(r * N)];
                op[(unsigned)(r * N)] = v;
            }
        }
    }
}

// ---------------------------------------------------------------------------
// Banded (+/-3) multi-head attention. One wave per (b,h,t); lanes cover DK/4.
// Q,K,V: f32 [B*T, D] (row = b*T+t, head h at cols h*128..). Output bf16.
// ---------------------------------------------------------------------------
__global__ __launch_bounds__(256)
void attn_banded_kernel(const float* __restrict__ Q,
                        const float* __restrict__ K,
                        const float* __restrict__ V,
                        unsigned short* __restrict__ Aout)
{
    const unsigned widx = blockIdx.x * 8 + (threadIdx.x >> 5);
    const unsigned lane = threadIdx.x & 31;
    const int t = widx % TT;
    const int h = (widx / TT) % HH;
    const int b = widx / (TT * HH);
    const unsigned base = (unsigned)(b * TT + t) * DD + h * DK + lane * 4;

    const float4 q = *(const float4*)(Q + base);
    float s[7];
#pragma unroll
    for (int jj = 0; jj < 7; ++jj) {
        const int t2 = t - 3 + jj;
        const bool ok = (t2 >= 0) && (t2 < TT);
        float p = 0.f;
        if (ok) {
            const unsigned kb = (unsigned)(b * TT + t2) * DD + h * DK + lane * 4;
            const float4 k4 = *(const float4*)(K + kb);
            p = q.x * k4.x + q.y * k4.y + q.z * k4.z + q.w * k4.w;
        }
#pragma unroll
        for (int off = 16; off > 0; off >>= 1)
            p += __shfl_xor(p, off, 32);
        s[jj] = ok ? p * 0.08838834764831845f : -1e10f;   // 1/sqrt(128)
    }
    float m = s[0];
#pragma unroll
    for (int jj = 1; jj < 7; ++jj) m = fmaxf(m, s[jj]);
    float wgt[7], den = 0.f;
#pragma unroll
    for (int jj = 0; jj < 7; ++jj) { wgt[jj] = __expf(s[jj] - m); den += wgt[jj]; }
    const float inv = 1.f / den;

    float ox = 0.f, oy = 0.f, oz = 0.f, ow = 0.f;
#pragma unroll
    for (int jj = 0; jj < 7; ++jj) {
        const int t2 = t - 3 + jj;
        if (t2 >= 0 && t2 < TT) {
            const unsigned vb = (unsigned)(b * TT + t2) * DD + h * DK + lane * 4;
            const float4 v4 = *(const float4*)(V + vb);
            const float wj = wgt[jj] * inv;
            ox += wj * v4.x; oy += wj * v4.y; oz += wj * v4.z; ow += wj * v4.w;
        }
    }
    // pack 4 bf16 -> 8 bytes
    uint2 pk;
    pk.x = (unsigned)f2bf(ox) | ((unsigned)f2bf(oy) << 16);
    pk.y = (unsigned)f2bf(oz) | ((unsigned)f2bf(ow) << 16);
    *(uint2*)(Aout + base) = pk;
}

// ---------------------------------------------------------------------------
// BiGRU recurrence. grid = 8 blocks: dir = blk>>2, batch group g = blk&3
// (16 batches per group). h[16,256] kept in LDS (f32 master + bf16 WMMA copy).
// Per step each wave computes a 32-col strip of all 3 gates (r/z/n) with
// v_wmma_f32_16x16x32_bf16 on h_bf x W_hh^T, then does gate math in-register.
// ---------------------------------------------------------------------------
__global__ __launch_bounds__(256)
void gru_kernel(const float* __restrict__ gxF, const float* __restrict__ gxB,
                const unsigned short* __restrict__ WhhF,
                const unsigned short* __restrict__ WhhB,
                const float* __restrict__ bhhF, const float* __restrict__ bhhB,
                const int* __restrict__ lens, float* __restrict__ out)
{
    __shared__ unsigned short h_bf[16 * GHID];   // 8 KB
    __shared__ float          h_f [16 * GHID];   // 16 KB
    __shared__ int            Ls[16];

    const unsigned tid = threadIdx.x;
    const unsigned dir = blockIdx.x >> 2;
    const unsigned g   = blockIdx.x & 3;
    const float* gx           = dir ? gxB  : gxF;
    const unsigned short* Whh = dir ? WhhB : WhhF;
    const float* bhh          = dir ? bhhB : bhhF;

    for (unsigned idx = tid; idx < 16 * GHID; idx += 256) { h_f[idx] = 0.f; h_bf[idx] = 0; }
    if (tid < 16) Ls[tid] = lens[g * 16 + tid];
    __syncthreads();

    const unsigned w    = tid >> 5;
    const unsigned lane = tid & 31;
    const unsigned lr   = lane & 15;
    const unsigned hi   = lane >> 4;
    const unsigned cbase = w * 32;    // 8 waves x 32 cols = GH=256

    for (int t = 0; t < TT; ++t) {
        v8f acc[3][2];
#pragma unroll
        for (int gi = 0; gi < 3; ++gi)
#pragma unroll
            for (int j = 0; j < 2; ++j)
                acc[gi][j] = (v8f){0.f,0.f,0.f,0.f,0.f,0.f,0.f,0.f};

        for (int k0 = 0; k0 < GHID; k0 += 32) {
            const unsigned kc = (unsigned)k0 + hi * 8;
            FragU fa;
            fa.u[0] = *(const u32x4*)&h_bf[lr * GHID + kc];
            fa.u[1] = *(const u32x4*)&h_bf[lr * GHID + kc + 16];
            const v16bf a = fa.v;
#pragma unroll
            for (int gi = 0; gi < 3; ++gi)
#pragma unroll
                for (int j = 0; j < 2; ++j) {
                    const unsigned n = gi * GHID + cbase + j * 16 + lr;
                    const v16bf bfr = load_frag_g(Whh + n * (unsigned)GHID + kc);
                    acc[gi][j] = __builtin_amdgcn_wmma_f32_16x16x32_bf16(
                        false, a, false, bfr, (short)0, acc[gi][j], false, false);
                }
        }
        __syncthreads();   // all waves finished reading h_bf

        // gate update: lane element (j, r) -> (row i = hi*8+r, col c = cbase+j*16+lr)
#pragma unroll
        for (int j = 0; j < 2; ++j) {
            const unsigned c  = cbase + j * 16 + lr;
            const float br = bhh[c], bz = bhh[GHID + c], bn = bhh[2 * GHID + c];
#pragma unroll
            for (int r = 0; r < 8; ++r) {
                const unsigned i = hi * 8 + r;
                const int L = Ls[i];
                if (t < L) {
                    const unsigned bb  = g * 16 + i;
                    const int st  = dir ? (L - 1 - t) : t;
                    const unsigned gro = (bb * TT + (unsigned)st) * (3 * GHID);
                    const float xr = gx[gro + c];
                    const float xz = gx[gro + GHID + c];
                    const float xn = gx[gro + 2 * GHID + c];
                    const float hr = acc[0][j][r] + br;
                    const float hz = acc[1][j][r] + bz;
                    const float hn = acc[2][j][r] + bn;
                    const float rr = 1.f / (1.f + __expf(-(xr + hr)));
                    const float zz = 1.f / (1.f + __expf(-(xz + hz)));
                    const float nn = tanhf(xn + rr * hn);
                    const float hp = h_f[i * GHID + c];
                    const float hnew = (1.f - zz) * nn + zz * hp;
                    h_f[i * GHID + c]  = hnew;
                    h_bf[i * GHID + c] = f2bf(hnew);
                    out[(bb * TT + (unsigned)st) * HID + dir * GHID + c] = hnew;
                }
            }
        }
        __syncthreads();   // h updated before next step's WMMA reads
    }
}

// ---------------------------------------------------------------------------
// Host-side orchestration
// ---------------------------------------------------------------------------
extern "C" void kernel_launch(void* const* d_in, const int* in_sizes, int n_in,
                              void* d_out, int out_size, void* d_ws, size_t ws_size,
                              hipStream_t stream) {
    (void)in_sizes; (void)n_in; (void)ws_size;

    const float* seg    = (const float*)d_in[0];
    const int*   seglen = (const int*)d_in[1];
    const float* Wq = (const float*)d_in[2];  const float* bq = (const float*)d_in[3];
    const float* Wk = (const float*)d_in[4];  const float* bk = (const float*)d_in[5];
    const float* Wv = (const float*)d_in[6];  const float* bv = (const float*)d_in[7];
    const float* Wo = (const float*)d_in[8];  const float* bo = (const float*)d_in[9];
    const float* Wihf = (const float*)d_in[10]; const float* Whhf = (const float*)d_in[11];
    const float* bihf = (const float*)d_in[12]; const float* bhhf = (const float*)d_in[13];
    const float* Wihb = (const float*)d_in[14]; const float* Whhb = (const float*)d_in[15];
    const float* bihb = (const float*)d_in[16]; const float* bhhb = (const float*)d_in[17];

    // workspace carve-up
    char* p = (char*)d_ws;
    auto carve = [&](size_t bytes) { char* r = p; p += (bytes + 255) & ~(size_t)255; return r; };
    unsigned short* Xb   = (unsigned short*)carve((size_t)MTOT * DD * 2);   // bf16 stream
    float*          Xres = (float*)carve((size_t)MTOT * DD * 4);            // f32 residual
    float*          Qf   = (float*)carve((size_t)MTOT * DD * 4);            // also GXf
    float*          Kf   = (float*)carve((size_t)MTOT * DD * 4);            // also GXb
    float*          Vf   = (float*)carve((size_t)MTOT * DD * 4);
    unsigned short* Aout = (unsigned short*)carve((size_t)MTOT * DD * 2);
    unsigned short* wq_bf = (unsigned short*)carve((size_t)2 * DD * DD * 2);
    unsigned short* wk_bf = (unsigned short*)carve((size_t)2 * DD * DD * 2);
    unsigned short* wv_bf = (unsigned short*)carve((size_t)2 * DD * DD * 2);
    unsigned short* wo_bf = (unsigned short*)carve((size_t)2 * DD * DD * 2);
    unsigned short* wihf_bf = (unsigned short*)carve((size_t)3 * GHID * DD * 2);
    unsigned short* wihb_bf = (unsigned short*)carve((size_t)3 * GHID * DD * 2);
    unsigned short* whhf_bf = (unsigned short*)carve((size_t)3 * GHID * GHID * 2);
    unsigned short* whhb_bf = (unsigned short*)carve((size_t)3 * GHID * GHID * 2);

    auto cvt = [&](const float* src, unsigned short* dst, int n) {
        cvt_f32_bf16_kernel<<<1024, 256, 0, stream>>>(src, dst, n / 4);
    };

    // 1. convert all weights + the input stream to bf16
    cvt(Wq, wq_bf, 2 * DD * DD);
    cvt(Wk, wk_bf, 2 * DD * DD);
    cvt(Wv, wv_bf, 2 * DD * DD);
    cvt(Wo, wo_bf, 2 * DD * DD);
    cvt(Wihf, wihf_bf, 3 * GHID * DD);
    cvt(Wihb, wihb_bf, 3 * GHID * DD);
    cvt(Whhf, whhf_bf, 3 * GHID * GHID);
    cvt(Whhb, whhb_bf, 3 * GHID * GHID);
    cvt(seg, Xb, MTOT * DD);

    const dim3 gD(DD / 128, MTOT / 128);    // N=1024 GEMMs
    const dim3 gG(768 / 128, MTOT / 128);   // N=768 GEMMs

    // 2. two MHA layers
    for (int l = 0; l < 2; ++l) {
        const unsigned short* wq_l = wq_bf + (size_t)l * DD * DD;
        const unsigned short* wk_l = wk_bf + (size_t)l * DD * DD;
        const unsigned short* wv_l = wv_bf + (size_t)l * DD * DD;
        const unsigned short* wo_l = wo_bf + (size_t)l * DD * DD;
        const float* bq_l = bq + l * DD;
        const float* bk_l = bk + l * DD;
        const float* bv_l = bv + l * DD;
        const float* bo_l = bo + l * DD;

        gemm_bf16_kernel<false><<<gD, 256, 0, stream>>>(Xb, wq_l, bq_l, nullptr, Qf, MTOT, DD, DD);
        gemm_bf16_kernel<false><<<gD, 256, 0, stream>>>(Xb, wk_l, bk_l, nullptr, Kf, MTOT, DD, DD);
        gemm_bf16_kernel<false><<<gD, 256, 0, stream>>>(Xb, wv_l, bv_l, nullptr, Vf, MTOT, DD, DD);

        attn_banded_kernel<<<(BB * HH * TT) / 8, 256, 0, stream>>>(Qf, Kf, Vf, Aout);

        const float* resid = (l == 0) ? seg : (const float*)Xres;
        gemm_bf16_kernel<true><<<gD, 256, 0, stream>>>(Aout, wo_l, bo_l, resid, Xres, MTOT, DD, DD);
        cvt(Xres, Xb, MTOT * DD);
    }

    // 3. GRU input projections (reuse Qf/Kf as GXf/GXb)
    gemm_bf16_kernel<false><<<gG, 256, 0, stream>>>(Xb, wihf_bf, bihf, nullptr, Qf, MTOT, 768, DD);
    gemm_bf16_kernel<false><<<gG, 256, 0, stream>>>(Xb, wihb_bf, bihb, nullptr, Kf, MTOT, 768, DD);

    // 4. pre-zero output (invalid positions must read 0), then recurrence
    zero_f32_kernel<<<2048, 256, 0, stream>>>((float*)d_out, out_size);
    gru_kernel<<<8, 256, 0, stream>>>(Qf, Kf, whhf_bf, whhb_bf, bhhf, bhhb, seglen, (float*)d_out);
}
